// Attention_63883343561180
// MI455X (gfx1250) — matrix operand order repaired and checked by reading, compiled, not run on hardware
//
#include <hip/hip_runtime.h>
#include <hip/hip_bf16.h>

// ---------------------------------------------------------------------------
// Additive (Bahdanau) attention, MI455X / gfx1250, wave32 + WMMA bf16.
//   B=64, N=196, D1=512, D2=2048, H=512
// v2: LDS-staged bf16 A-tiles (convert-once, ds_load_b128 fragments),
//     M-tile=32 in the fused score kernel to halve W2 L2 traffic.
// ---------------------------------------------------------------------------

typedef __attribute__((ext_vector_type(2)))  float  v2f;
typedef __attribute__((ext_vector_type(4)))  float  v4f;
typedef __attribute__((ext_vector_type(8)))  float  v8f;
typedef __attribute__((ext_vector_type(16))) __bf16 v16bf;
typedef __attribute__((ext_vector_type(8)))  unsigned short v8u;
typedef __attribute__((ext_vector_type(16))) unsigned short v16u;

#define B_  64
#define N_  196
#define D1_ 512
#define D2_ 2048
#define H_  512
#define M_  (B_ * N_)        // 12544 = 392 * 32

#define KCHUNK 512
#define LDS_PITCH (KCHUNK + 8)   // +8 bf16 => row stride 1040B => bank rotate 4

// float -> bf16 round-to-nearest-even (bit trick)
static __device__ __forceinline__ unsigned short f2bf(float f) {
  unsigned u = __builtin_bit_cast(unsigned, f);
  unsigned r = (u + 0x7FFFu + ((u >> 16) & 1u)) >> 16;
  return (unsigned short)r;
}

__global__ __launch_bounds__(256) void cvt_f32_bf16(const float* __restrict__ src,
                                                    unsigned short* __restrict__ dst,
                                                    int n) {
  int i = blockIdx.x * blockDim.x + threadIdx.x;
  if (i < n) dst[i] = f2bf(src[i]);
}

static __device__ __forceinline__ v16bf combine16(v8u lo, v8u hi) {
  v16u u;
#pragma unroll
  for (int j = 0; j < 8; ++j) { u[j] = lo[j]; u[8 + j] = hi[j]; }
  return __builtin_bit_cast(v16bf, u);
}

// A-fragment (16x32 bf16) from an LDS row: element j<8 -> K=kb+j,
// j>=8 -> K=kb+16+(j-8), kb = k0 + (lane>>4)*8.
static __device__ __forceinline__ v16bf load_a_lds(const unsigned short* __restrict__ rowp,
                                                   int kb) {
  v8u lo = *(const v8u*)(rowp + kb);
  v8u hi = *(const v8u*)(rowp + kb + 16);
  return combine16(lo, hi);
}

// B-fragment (32x16 bf16): lane l holds column n=l%16; element j<8 -> K=kb+j,
// j>=8 -> K=kb+16+(j-8), kb = k0 + (lane>>4)*8 (row stride H_).
static __device__ __forceinline__ v16bf load_b_bf16(const unsigned short* __restrict__ p) {
  v16u u;
#pragma unroll
  for (int j = 0; j < 8; ++j) { u[j] = p[j * H_]; u[8 + j] = p[(16 + j) * H_]; }
  return __builtin_bit_cast(v16bf, u);
}

// Cooperative fp32 -> bf16 stage of ROWS x KCHUNK tile into LDS.
// Each thread converts 8 contiguous elements per iteration (coalesced b128s).
template <int ROWS>
static __device__ __forceinline__ void stage_tile(const float* __restrict__ gbase, // [.. , ld]
                                                  size_t ld, int kc,
                                                  unsigned short* __restrict__ sA,
                                                  int tid) {
  constexpr int ITERS = ROWS * KCHUNK / (256 * 8);
#pragma unroll
  for (int it = 0; it < ITERS; ++it) {
    int e   = it * 2048 + tid * 8;
    int row = e >> 9;          // /KCHUNK
    int col = e & (KCHUNK - 1);
    const float* p = gbase + (size_t)row * ld + kc + col;
    v4f f0 = *(const v4f*)p;
    v4f f1 = *(const v4f*)(p + 4);
    v8u o;
#pragma unroll
    for (int j = 0; j < 4; ++j) { o[j] = f2bf(f0[j]); o[4 + j] = f2bf(f1[j]); }
    *(v8u*)(sA + row * LDS_PITCH + col) = o;
  }
}

// --------------------------- enc1 = input1 @ W1 + b1 -----------------------
// grid = B/16 workgroups of 8 waves; wave w owns columns [w*64, w*64+64).
__global__ __launch_bounds__(256) void enc1_kernel(const float* __restrict__ in1,
                                                   const unsigned short* __restrict__ W1bf,
                                                   const float* __restrict__ b1,
                                                   float* __restrict__ enc1) {
  __shared__ __align__(16) unsigned short sA[16 * LDS_PITCH];
  const int tid  = threadIdx.x;
  const int lane = tid & 31;
  const int wave = tid >> 5;
  const int m0   = blockIdx.x * 16;
  const int half = lane >> 4;
  const int ln   = lane & 15;
  const int hb   = wave * 64;

  stage_tile<16>(in1 + (size_t)m0 * D1_, D1_, 0, sA, tid);
  __syncthreads();

  const unsigned short* sArow = sA + ln * LDS_PITCH;
  v8f c[4] = {};
  for (int k0 = 0; k0 < D1_; k0 += 32) {
    v16bf a = load_a_lds(sArow, k0 + half * 8);
#pragma unroll
    for (int t = 0; t < 4; ++t) {
      const unsigned short* bp = W1bf + (size_t)(k0 + half * 8) * H_ + hb + t * 16 + ln;
      v16bf bfrag = load_b_bf16(bp);
      c[t] = __builtin_amdgcn_wmma_f32_16x16x32_bf16(false, a, false, bfrag,
                                                     (short)0, c[t], false, false);
    }
  }
#pragma unroll
  for (int t = 0; t < 4; ++t) {
    int   h    = hb + t * 16 + ln;
    float bias = b1[h];
#pragma unroll
    for (int v = 0; v < 8; ++v) {
      int m = m0 + v + 8 * half;               // C/D layout: row = v + 8*half
      enc1[(size_t)m * H_ + h] = c[t][v] + bias;
    }
  }
}

// ---- scores[m] = relu(input2[m]@W2 + b2 + enc1[b(m)]) . Wf  (fused) -------
// grid = M/32 = 392 WGs; each computes 32 rows x 512 cols of enc2 in
// registers (never stored), reducing over H into 32 scalar scores.
__global__ __launch_bounds__(256) void score_kernel(const float* __restrict__ in2,
                                                    const unsigned short* __restrict__ W2bf,
                                                    const float* __restrict__ b2,
                                                    const float* __restrict__ enc1,
                                                    const float* __restrict__ Wf,
                                                    float* __restrict__ scores) {
  __shared__ __align__(16) unsigned short sA[32 * LDS_PITCH];  // 33.3 KB
  __shared__ float sscore[32];

  const int tid  = threadIdx.x;
  if (tid < 32) sscore[tid] = 0.0f;

  const int lane = tid & 31;
  const int wave = tid >> 5;
  const int m0   = blockIdx.x * 32;
  const int half = lane >> 4;
  const int ln   = lane & 15;
  const int hb   = wave * 64;

  const unsigned short* sArow0 = sA + ln * LDS_PITCH;
  const unsigned short* sArow1 = sA + (16 + ln) * LDS_PITCH;

  v8f c0[4] = {};
  v8f c1[4] = {};
  for (int kc = 0; kc < D2_; kc += KCHUNK) {
    __syncthreads();   // previous chunk fully consumed
    stage_tile<32>(in2 + (size_t)m0 * D2_, D2_, kc, sA, tid);
    __syncthreads();
#pragma unroll 2
    for (int k0 = 0; k0 < KCHUNK; k0 += 32) {
      int kb = k0 + half * 8;
      v16bf a0 = load_a_lds(sArow0, kb);
      v16bf a1 = load_a_lds(sArow1, kb);
      const unsigned short* bcol = W2bf + (size_t)(kc + k0 + half * 8) * H_ + hb + ln;
#pragma unroll
      for (int t = 0; t < 4; ++t) {
        v16bf bfrag = load_b_bf16(bcol + t * 16);
        c0[t] = __builtin_amdgcn_wmma_f32_16x16x32_bf16(false, a0, false, bfrag,
                                                        (short)0, c0[t], false, false);
        c1[t] = __builtin_amdgcn_wmma_f32_16x16x32_bf16(false, a1, false, bfrag,
                                                        (short)0, c1[t], false, false);
      }
    }
  }

  // epilogue: +b2 +enc1[b], relu, .Wf, reduce over this wave's 64 columns
  float part[2][8];
#pragma unroll
  for (int rt = 0; rt < 2; ++rt)
#pragma unroll
    for (int v = 0; v < 8; ++v) part[rt][v] = 0.0f;

#pragma unroll
  for (int t = 0; t < 4; ++t) {
    int   h    = hb + t * 16 + ln;
    float bias = b2[h];
    float wf   = Wf[h];
#pragma unroll
    for (int v = 0; v < 8; ++v) {
      {
        int   m = m0 + v + 8 * half;
        int   b = m / N_;
        float x = c0[t][v] + bias + enc1[(size_t)b * H_ + h];
        part[0][v] += fmaxf(x, 0.0f) * wf;
      }
      {
        int   m = m0 + 16 + v + 8 * half;
        int   b = m / N_;
        float x = c1[t][v] + bias + enc1[(size_t)b * H_ + h];
        part[1][v] += fmaxf(x, 0.0f) * wf;
      }
    }
  }
  // butterfly over the 16 lanes that share a row (masks 1/2/4/8 stay in-half)
#pragma unroll
  for (int rt = 0; rt < 2; ++rt)
#pragma unroll
    for (int v = 0; v < 8; ++v) {
      float p = part[rt][v];
      p += __shfl_xor(p, 1, 32);
      p += __shfl_xor(p, 2, 32);
      p += __shfl_xor(p, 4, 32);
      p += __shfl_xor(p, 8, 32);
      if (ln == 0) atomicAdd(&sscore[rt * 16 + v + 8 * half], p);
    }
  __syncthreads();
  if (tid < 32) scores[m0 + tid] = sscore[tid];
}

// --------------------------- softmax over N per batch ----------------------
__global__ __launch_bounds__(256) void softmax_kernel(const float* __restrict__ scores,
                                                      const float* __restrict__ bfp,
                                                      float* __restrict__ alpha) {
  __shared__ float sh[256];
  const int b = blockIdx.x, tid = threadIdx.x;
  float s = (tid < N_) ? (scores[b * N_ + tid] + bfp[0]) : -3.4e38f;
  sh[tid] = s;
  __syncthreads();
  for (int o = 128; o > 0; o >>= 1) {
    if (tid < o) sh[tid] = fmaxf(sh[tid], sh[tid + o]);
    __syncthreads();
  }
  float mx = sh[0];
  __syncthreads();
  float e = (tid < N_) ? __expf(s - mx) : 0.0f;
  sh[tid] = e;
  __syncthreads();
  for (int o = 128; o > 0; o >>= 1) {
    if (tid < o) sh[tid] += sh[tid + o];
    __syncthreads();
  }
  float inv = 1.0f / sh[0];
  if (tid < N_) alpha[b * N_ + tid] = e * inv;
}

// ----------------- att[b,:] = sum_n alpha[b,n] * input2[b,n,:] -------------
// grid = B*4; each WG handles 512 columns with float2 streaming loads.
__global__ __launch_bounds__(256) void att_kernel(const float* __restrict__ in2,
                                                  const float* __restrict__ alpha,
                                                  float* __restrict__ att) {
  __shared__ float sa[N_];
  const int b     = blockIdx.x >> 2;
  const int chunk = blockIdx.x & 3;
  const int tid   = threadIdx.x;
  if (tid < N_) sa[tid] = alpha[b * N_ + tid];
  __syncthreads();

  const int d = chunk * 512 + tid * 2;
  const float* base = in2 + (size_t)b * N_ * D2_ + d;
  v2f acc = {0.0f, 0.0f};
  for (int n = 0; n < N_; ++n) {
    float a = sa[n];
    v2f   x = *(const v2f*)(base + (size_t)n * D2_);
    acc += a * x;
  }
  *(v2f*)(att + (size_t)b * D2_ + d) = acc;
}

// ---------------------------------------------------------------------------
extern "C" void kernel_launch(void* const* d_in, const int* in_sizes, int n_in,
                              void* d_out, int out_size, void* d_ws, size_t ws_size,
                              hipStream_t stream) {
  (void)in_sizes; (void)n_in; (void)out_size; (void)ws_size;
  const float* input1 = (const float*)d_in[0];
  const float* input2 = (const float*)d_in[1];
  const float* W1     = (const float*)d_in[2];
  const float* b1     = (const float*)d_in[3];
  const float* W2     = (const float*)d_in[4];
  const float* b2     = (const float*)d_in[5];
  const float* Wf     = (const float*)d_in[6];
  const float* bfp    = (const float*)d_in[7];

  float* out   = (float*)d_out;
  float* att   = out;              // [64, 2048]
  float* alpha = out + B_ * D2_;   // [64, 196]

  char* ws = (char*)d_ws;
  unsigned short* W2bf  = (unsigned short*)ws;                                  // 2 MB
  unsigned short* W1bf  = (unsigned short*)(ws + 2097152);                      // 512 KB
  float*          enc1  = (float*)(ws + 2097152 + 524288);                      // 128 KB
  float*          score = (float*)(ws + 2097152 + 524288 + 131072);             // 49 KB

  cvt_f32_bf16<<<(D2_ * H_ + 255) / 256, 256, 0, stream>>>(W2, W2bf, D2_ * H_);
  cvt_f32_bf16<<<(D1_ * H_ + 255) / 256, 256, 0, stream>>>(W1, W1bf, D1_ * H_);
  enc1_kernel <<<B_ / 16, 256, 0, stream>>>(input1, W1bf, b1, enc1);
  score_kernel<<<M_ / 32, 256, 0, stream>>>(input2, W2bf, b2, enc1, Wf, score);
  softmax_kernel<<<B_, 256, 0, stream>>>(score, bfp, alpha);
  att_kernel  <<<B_ * 4, 256, 0, stream>>>(input2, alpha, att);
}
